// Transducer_798863917292
// MI455X (gfx1250) — compile-verified, matrix operand order checked by reading, not run on hardware
//
#include <hip/hip_runtime.h>
#include <math.h>

// ---------------------------------------------------------------------------
// RNN-T loss for MI455X (gfx1250, wave32).
// Phase 1 (HBM-bound, ~23us roofline): streaming log-softmax stats; row sums
//          accumulated on the matrix pipe (v_wmma_f32_16x16x32_f16, B = ones)
//          with online-max rescaling so VALU(exp) and XDL(sum) co-execute.
// Phase 2 (latency-bound): single wave per batch element runs the 383-step
//          anti-diagonal logaddexp DP entirely in registers (shuffle-only).
// Phase 3: scalar mean.
// ---------------------------------------------------------------------------

typedef __attribute__((ext_vector_type(16))) _Float16 v16h;
typedef __attribute__((ext_vector_type(8)))  float    v8f;

#define RNNT_B   8
#define RNNT_T   256
#define RNNT_U   128            // U_MAX
#define RNNT_U1  129            // U_MAX + 1
#define RNNT_V   512
#define RNNT_NROWS   (RNNT_B * RNNT_T * RNNT_U1)   // 264192, divisible by 16
#define RNNT_NGROUPS (RNNT_NROWS / 16)             // 16512
#define NEG_INF (-__builtin_inff())

__device__ __forceinline__ float lse2(float a, float b) {
  float m = fmaxf(a, b);
  if (m == NEG_INF) return NEG_INF;
  return m + __logf(__expf(a - m) + __expf(b - m));
}

// ---------------------------------------------------------------------------
// Kernel 1: per-row logZ via single-pass streaming; one wave = 16 rows laid
// out directly in WMMA A-operand order. Emits blank/emit log-probs.
// ---------------------------------------------------------------------------
__global__ __launch_bounds__(256) void rnnt_softmax_stats(
    const float* __restrict__ logits, const int* __restrict__ y,
    float* __restrict__ blank_lp, float* __restrict__ emit_lp)
{
  const int lane = threadIdx.x & 31;
  const int wave = threadIdx.x >> 5;
  const int group = blockIdx.x * 8 + wave;            // grid sized exactly
  const int half = lane >> 4;                          // 0: even chunk, 1: odd
  const int r    = lane & 15;                          // row within group
  const size_t rowBase = (size_t)group * 16;
  const float* __restrict__ rowPtr = logits + (rowBase + (size_t)r) * RNNT_V;

  const v16h ones = { (_Float16)1.f,(_Float16)1.f,(_Float16)1.f,(_Float16)1.f,
                      (_Float16)1.f,(_Float16)1.f,(_Float16)1.f,(_Float16)1.f,
                      (_Float16)1.f,(_Float16)1.f,(_Float16)1.f,(_Float16)1.f,
                      (_Float16)1.f,(_Float16)1.f,(_Float16)1.f,(_Float16)1.f };
  v8f acc = {};               // f32 row-sum accumulators (C/D of WMMA)
  float m = NEG_INF;          // running row max (synchronized across lane pair)

  for (int k = 0; k < 16; ++k) {
    // Lane pair (r, r+16) covers columns [32k, 32k+32) of row r; each lane
    // loads its 8+8 columns matching the 16-bit A-matrix VGPR layout.
    const float* p = rowPtr + k * 32 + half * 8;
    float4 x0 = *(const float4*)(p + 0);
    float4 x1 = *(const float4*)(p + 4);
    float4 x2 = *(const float4*)(p + 16);
    float4 x3 = *(const float4*)(p + 20);
    if (k + 1 < 16) __builtin_prefetch(p + 32, 0, 0);  // global_prefetch_b8

    // chunk max -> per-row max (combine the lane pair), then online rescale
    float cm = fmaxf(fmaxf(fmaxf(x0.x, x0.y), fmaxf(x0.z, x0.w)),
               fmaxf(fmaxf(fmaxf(x1.x, x1.y), fmaxf(x1.z, x1.w)),
               fmaxf(fmaxf(fmaxf(x2.x, x2.y), fmaxf(x2.z, x2.w)),
                     fmaxf(fmaxf(x3.x, x3.y), fmaxf(x3.z, x3.w)))));
    cm = fmaxf(cm, __shfl_xor(cm, 16, 32));
    float mnew = fmaxf(m, cm);
    float scale = __expf(m - mnew);       // 0 on first chunk, 1 if unchanged
    // C layout: acc[j] = row j (lanes 0-15) / row j+8 (lanes 16-31)
    #pragma unroll
    for (int j = 0; j < 8; ++j)
      acc[j] *= __shfl(scale, j + half * 8, 32);
    m = mnew;

    v16h a;
    a[0]  = (_Float16)__expf(x0.x - m);  a[1]  = (_Float16)__expf(x0.y - m);
    a[2]  = (_Float16)__expf(x0.z - m);  a[3]  = (_Float16)__expf(x0.w - m);
    a[4]  = (_Float16)__expf(x1.x - m);  a[5]  = (_Float16)__expf(x1.y - m);
    a[6]  = (_Float16)__expf(x1.z - m);  a[7]  = (_Float16)__expf(x1.w - m);
    a[8]  = (_Float16)__expf(x2.x - m);  a[9]  = (_Float16)__expf(x2.y - m);
    a[10] = (_Float16)__expf(x2.z - m);  a[11] = (_Float16)__expf(x2.w - m);
    a[12] = (_Float16)__expf(x3.x - m);  a[13] = (_Float16)__expf(x3.y - m);
    a[14] = (_Float16)__expf(x3.z - m);  a[15] = (_Float16)__expf(x3.w - m);

    // row sums on the matrix pipe: D[m][n] = sum_k exp(...) for every n
    acc = __builtin_amdgcn_wmma_f32_16x16x32_f16(
        false, a, false, ones, (short)0, acc, false, false);
  }

  // Lanes 0-7 hold sums of rows 0-7 in acc[0..7]; lanes 16-31 hold rows 8-15.
  const bool active = (lane < 8) || (lane >= 16 && lane < 24);
  const int  rloc = half ? (lane - 8) : lane;     // row this lane finalizes
  const int  j = rloc & 7;
  float s = acc[0];
  if (j == 1) s = acc[1];
  if (j == 2) s = acc[2];
  if (j == 3) s = acc[3];
  if (j == 4) s = acc[4];
  if (j == 5) s = acc[5];
  if (j == 6) s = acc[6];
  if (j == 7) s = acc[7];
  float mr = __shfl(m, rloc, 32);                 // m(row) lives in lane=row

  if (active) {
    const size_t row = rowBase + (size_t)rloc;
    const float logZ = mr + __logf(s);
    const unsigned u  = (unsigned)(row % RNNT_U1);
    const size_t   bt = row / RNNT_U1;            // b*T + t
    const float* rp = logits + row * RNNT_V;
    blank_lp[row] = rp[0] - logZ;
    if (u < RNNT_U) {
      const unsigned b = (unsigned)(bt / RNNT_T);
      emit_lp[bt * RNNT_U + u] = rp[y[b * RNNT_U + u]] - logZ;
    }
  }
}

// ---------------------------------------------------------------------------
// Kernel 2: one wave per batch element; anti-diagonal DP, 129 cells held as
// 5 register slots/lane, left-neighbor via shuffles. No LDS, no barriers.
// ---------------------------------------------------------------------------
__global__ __launch_bounds__(32) void rnnt_alpha(
    const float* __restrict__ blank_lp, const float* __restrict__ emit_lp,
    const int* __restrict__ T_len, const int* __restrict__ U_len,
    float* __restrict__ partial)
{
  const int b = blockIdx.x;
  const int lane = threadIdx.x;
  const float* __restrict__ BL = blank_lp + (size_t)b * RNNT_T * RNNT_U1;
  const float* __restrict__ EM = emit_lp  + (size_t)b * RNNT_T * RNNT_U;
  const int Tl = T_len[b];
  const int Ul = U_len[b];

  float prev[5], cur[5];
  #pragma unroll
  for (int s = 0; s < 5; ++s) prev[s] = NEG_INF;
  if (lane == 0) prev[0] = 0.0f;                  // alpha(0,0) = 0

  float saved = 0.0f;
  if (lane == 0 && Tl == 1 && Ul == 0) saved = BL[0];

  for (int d = 1; d <= (RNNT_T - 1) + RNNT_U; ++d) {
    float left[5];
    #pragma unroll
    for (int s = 0; s < 5; ++s) {                 // alpha(t, u-1) from d-1
      float upA = __shfl_up(prev[s], 1, 32);
      float upB = NEG_INF;
      if (s > 0) upB = __shfl(prev[s - 1], 31, 32);
      left[s] = (lane == 0) ? upB : upA;
    }
    #pragma unroll
    for (int s = 0; s < 5; ++s) {
      const int u = lane + 32 * s;
      const int t = d - u;
      float v = NEG_INF;
      if (u <= RNNT_U && t >= 0 && t < RNNT_T) {
        float top = (t >= 1) ? prev[s] + BL[(t - 1) * RNNT_U1 + u] : NEG_INF;
        float lef = (u >= 1) ? left[s] + EM[t * RNNT_U + (u - 1)] : NEG_INF;
        v = lse2(top, lef);
        if (t == Tl - 1 && u == Ul)
          saved = v + BL[(Tl - 1) * RNNT_U1 + Ul];
      }
      cur[s] = v;
    }
    #pragma unroll
    for (int s = 0; s < 5; ++s) prev[s] = cur[s];
  }

  float res = __shfl(saved, Ul & 31, 32);         // owner lane = Ul % 32
  if (lane == 0) partial[b] = res;
}

// ---------------------------------------------------------------------------
// Kernel 3: negative mean of the 8 per-batch log-probs.
// ---------------------------------------------------------------------------
__global__ void rnnt_finalize(const float* __restrict__ partial,
                              float* __restrict__ out)
{
  float s = 0.0f;
  for (int i = 0; i < RNNT_B; ++i) s += partial[i];
  out[0] = -s / (float)RNNT_B;
}

extern "C" void kernel_launch(void* const* d_in, const int* in_sizes, int n_in,
                              void* d_out, int out_size, void* d_ws, size_t ws_size,
                              hipStream_t stream)
{
  (void)in_sizes; (void)n_in; (void)out_size; (void)ws_size;
  const float* logits = (const float*)d_in[0];
  const int*   y      = (const int*)d_in[1];
  const int*   T_len  = (const int*)d_in[2];
  const int*   U_len  = (const int*)d_in[3];

  // workspace: blank_lp[264192] | emit_lp[262144] | partial[8]  (~2.1 MB)
  float* blank_lp = (float*)d_ws;
  float* emit_lp  = blank_lp + RNNT_NROWS;
  float* partial  = emit_lp + (size_t)RNNT_B * RNNT_T * RNNT_U;
  float* out      = (float*)d_out;

  rnnt_softmax_stats<<<RNNT_NGROUPS / 8, 256, 0, stream>>>(logits, y,
                                                           blank_lp, emit_lp);
  rnnt_alpha<<<RNNT_B, 32, 0, stream>>>(blank_lp, emit_lp, T_len, U_len,
                                        partial);
  rnnt_finalize<<<1, 1, 0, stream>>>(partial, out);
}
